// EncoderBi_82918638616698
// MI455X (gfx1250) — compile-verified, matrix operand order checked by reading, not run on hardware
//
#include <hip/hip_runtime.h>
#include <stdint.h>

#define Bc 128
#define Tc 512
#define Ec 256
#define Hc 512
#define G4 (4*Hc)   // 2048 gate width

typedef __attribute__((ext_vector_type(16))) __bf16 v16bf;
typedef __attribute__((ext_vector_type(8)))  float  v8f;
typedef __attribute__((ext_vector_type(4)))  unsigned int v4u;
typedef __attribute__((ext_vector_type(4)))  int v4i;
typedef __attribute__((ext_vector_type(8)))  int v8i;

union FragU { v16bf v; uint4 q[2]; };
union AccU  { v8f v; float f[8]; };

__device__ __forceinline__ unsigned short f2bf(float x) {
  unsigned u = __builtin_bit_cast(unsigned, x);
  unsigned r = u + 0x7fffu + ((u >> 16) & 1u);   // round-to-nearest-even
  return (unsigned short)(r >> 16);
}

__device__ __forceinline__ float sigm(float x)  { return 1.f / (1.f + __expf(-x)); }
__device__ __forceinline__ float tanhx(float x) { return 1.f - 2.f / (__expf(2.f * x) + 1.f); }

// Load one 16x32 bf16 WMMA fragment (A-style layout, CDNA5 ISA 7.12.2):
// lanes 0-15: K=k0+0..7 and k0+16..23 ; lanes 16-31: K=k0+8..15 and k0+24..31.
// Works for B operand too since we store B transposed [N,K].
__device__ __forceinline__ v16bf load_frag(const unsigned short* base, int ldk, int k0, int lane) {
  const int lr = lane & 15, kh = lane >> 4;
  const unsigned short* p = base + (size_t)lr * ldk + k0 + kh * 8;
  FragU u;
  u.q[0] = *reinterpret_cast<const uint4*>(p);
  u.q[1] = *reinterpret_cast<const uint4*>(p + 16);
  return u.v;
}

#define WMMA_BF16(a, b, c) \
  __builtin_amdgcn_wmma_f32_16x16x32_bf16(false, (a), false, (b), (short)0, (c), false, false)

// ---------------- Tensor Data Mover: up-to-3D tile load Global -> LDS ----------------
// D# per CDNA5 ISA ch.8: group0 = {flags, lds_addr, global_addr(57b), type=2},
// group1 = {data_size=2B, tensor_dim0/1, tile_dim0/1/2, dim0/dim1 strides(48b)},
// group2 = {tensor_dim2}, group3 = 0. Elements are 2-byte (bf16).
__device__ __forceinline__ void tdm_load(unsigned ldsAddr, const void* gptr,
                                         unsigned td0, unsigned td1, unsigned td2,
                                         unsigned tile0, unsigned tile1, unsigned tile2,
                                         unsigned long long stride0,
                                         unsigned long long stride1) {
  unsigned long long ga = (unsigned long long)gptr;
  v4u g0 = { 1u,                                  // count=1, user mode
             ldsAddr,                             // LDS byte address
             (unsigned)(ga & 0xffffffffu),
             (unsigned)((ga >> 32) & 0x01ffffffull) | 0x80000000u }; // addr[56:32] | type=2
  v8i g1;
  g1[0] = (int)(1u << 16);                                              // data_size=1 -> 2 bytes
  g1[1] = (int)((td0 & 0xffffu) << 16);                                 // tensor_dim0 lo
  g1[2] = (int)(((td0 >> 16) & 0xffffu) | ((td1 & 0xffffu) << 16));     // dim0 hi | dim1 lo
  g1[3] = (int)(((td1 >> 16) & 0xffffu) | ((tile0 & 0xffffu) << 16));   // dim1 hi | tile0
  g1[4] = (int)((tile1 & 0xffffu) | ((tile2 & 0xffffu) << 16));         // tile1 | tile2
  g1[5] = (int)(stride0 & 0xffffffffull);                               // dim0_stride[31:0]
  g1[6] = (int)(((stride0 >> 32) & 0xffffull) | ((stride1 & 0xffffull) << 16));
  g1[7] = (int)((stride1 >> 16) & 0xffffffffull);                       // dim1_stride[47:16]
  v4i g2 = { (int)td2, 0, 0, 0 };   // tensor_dim2
  v4i g3 = { 0, 0, 0, 0 };
#if defined(__clang_major__) && __clang_major__ >= 23
  v8i z8 = { 0, 0, 0, 0, 0, 0, 0, 0 };
  __builtin_amdgcn_tensor_load_to_lds(g0, g1, g2, g3, z8, 0);
#else
  __builtin_amdgcn_tensor_load_to_lds(g0, g1, g2, g3, 0);
#endif
}

// ---------------- embedding gather + cast ----------------
__global__ __launch_bounds__(256) void k_embed(const int* __restrict__ x,
                                               const float* __restrict__ emb,
                                               unsigned short* __restrict__ e) {
  const int token = blockIdx.x;              // B*T tokens
  const int row = x[token];
  const int c = threadIdx.x;                 // E == 256
  e[(size_t)token * Ec + c] = f2bf(emb[(size_t)row * Ec + c]);
}

// ---------------- transpose + cast weights: f32 [K,N] -> bf16 [N,K] ----------------
__global__ __launch_bounds__(256) void k_transpose_cast(const float* __restrict__ in,
                                                        unsigned short* __restrict__ out,
                                                        int K, int N) {
  size_t idx = (size_t)blockIdx.x * 256 + threadIdx.x;
  if (idx >= (size_t)K * N) return;
  int k = (int)(idx / N), n = (int)(idx % N);
  out[(size_t)n * K + k] = f2bf(in[idx]);
}

// ---------------- zero h (double buffer) + c ----------------
__global__ __launch_bounds__(256) void k_zero_state(unsigned short* __restrict__ h0,
                                                    unsigned short* __restrict__ h1,
                                                    float* __restrict__ cc, int n) {
  int i = blockIdx.x * 256 + threadIdx.x;
  if (i < n) { h0[i] = 0; h1[i] = 0; cc[i] = 0.f; }
}

// ---------------- big input-projection GEMM: C[M,N] = A[M,K] * BT[N,K]^T + bias ----------------
// Dynamic LDS layout (base 0): As[2] @ 0 / 4096 (64x32 bf16), Bs[2] @ 8192 / 16384 (128x32 bf16).
// TDM double-buffered staging issued by wave 0; compute waves run pure ds_load + wmma.
__global__ __launch_bounds__(256) void k_gemm_bias(const unsigned short* __restrict__ A,
                                                   const unsigned short* __restrict__ BT,
                                                   const float* __restrict__ bias,
                                                   float* __restrict__ C,
                                                   int M, int N, int K) {
  extern __shared__ char smem[];
  const int tid = threadIdx.x;
  const int lane = tid & 31, w = tid >> 5;
  const int wm = w >> 2, wn = w & 3;                 // 2 x 4 wave grid
  const int m0b = blockIdx.y * 64, n0b = blockIdx.x * 128;
  const int m0w = wm * 32, n0w = wn * 32;
  const int nk = K >> 5;

  if (tid < 32) {  // wave 0 primes the pipeline
    tdm_load(0u,    A  + (size_t)m0b * K, (unsigned)K, (unsigned)M, 0, 32, 64,  0, (unsigned)K, 0);
    tdm_load(8192u, BT + (size_t)n0b * K, (unsigned)K, (unsigned)N, 0, 32, 128, 0, (unsigned)K, 0);
  }

  v8f acc00 = {0,0,0,0,0,0,0,0}, acc01 = {0,0,0,0,0,0,0,0};
  v8f acc10 = {0,0,0,0,0,0,0,0}, acc11 = {0,0,0,0,0,0,0,0};

  for (int i = 0; i < nk; ++i) {
    if (tid < 32) {
      if (i + 1 < nk) {
        const int k0n = (i + 1) << 5;
        const unsigned bufn = (unsigned)((i + 1) & 1);
        tdm_load(4096u * bufn,         A  + (size_t)m0b * K + k0n,
                 (unsigned)K, (unsigned)M, 0, 32, 64,  0, (unsigned)K, 0);
        tdm_load(8192u + 8192u * bufn, BT + (size_t)n0b * K + k0n,
                 (unsigned)K, (unsigned)N, 0, 32, 128, 0, (unsigned)K, 0);
        __builtin_amdgcn_s_wait_tensorcnt(2);   // current pair done, next pair in flight
      } else {
        __builtin_amdgcn_s_wait_tensorcnt(0);
      }
    }
    __syncthreads();
    const unsigned short* As = (const unsigned short*)(smem + 4096 * (i & 1));
    const unsigned short* Bs = (const unsigned short*)(smem + 8192 + 8192 * (i & 1));
    v16bf a0 = load_frag(&As[(m0w)      * 32], 32, 0, lane);
    v16bf a1 = load_frag(&As[(m0w + 16) * 32], 32, 0, lane);
    v16bf b0 = load_frag(&Bs[(n0w)      * 32], 32, 0, lane);
    v16bf b1 = load_frag(&Bs[(n0w + 16) * 32], 32, 0, lane);
    acc00 = WMMA_BF16(a0, b0, acc00);
    acc01 = WMMA_BF16(a0, b1, acc01);
    acc10 = WMMA_BF16(a1, b0, acc10);
    acc11 = WMMA_BF16(a1, b1, acc11);
    __syncthreads();
  }

  const int n = lane & 15, mb = (lane >> 4) * 8;
  const int cn0 = n0b + n0w + n, cn1 = cn0 + 16;
  const float bv0 = bias ? bias[cn0] : 0.f;
  const float bv1 = bias ? bias[cn1] : 0.f;
  AccU u00, u01, u10, u11;
  u00.v = acc00; u01.v = acc01; u10.v = acc10; u11.v = acc11;
#pragma unroll
  for (int r = 0; r < 8; ++r) {
    size_t mA = (size_t)(m0b + m0w + mb + r) * N;
    size_t mB = (size_t)(m0b + m0w + 16 + mb + r) * N;
    C[mA + cn0] = u00.f[r] + bv0;
    C[mA + cn1] = u01.f[r] + bv1;
    C[mB + cn0] = u10.f[r] + bv0;
    C[mB + cn1] = u11.f[r] + bv1;
  }
}

// ---------------- one LSTM timestep: z = xw_t + h * Wr ; gates ; c,h update ----------------
// Dynamic LDS layout (base 0):
//   Bls[2] @ 0 / 16384 : Wr strip [4 gates][64 rows][32 k] bf16 (3D TDM, z-dim = gate)
//   hA     @ 32768     : h tile [16][Hc] bf16 (2D TDM)
//   zs overlays buffer 0 after the k-loop (last chunk is in buffer 1; nk is even).
__global__ __launch_bounds__(256) void k_lstm_step(const float* __restrict__ xw_t,          // + t*G4
                                                   const unsigned short* __restrict__ WrT,  // [G4][Hc] bf16
                                                   const unsigned short* __restrict__ h_src,// [Bc][Hc] bf16
                                                   unsigned short* __restrict__ h_dst,
                                                   float* __restrict__ c,                   // [Bc][Hc]
                                                   float* __restrict__ seq_t) {             // + t*Hc
  extern __shared__ char smem[];
  const int tid = threadIdx.x, lane = tid & 31, w = tid >> 5;
  const int m0 = blockIdx.x * 16, j0 = blockIdx.y * 64;
  const int gate = w >> 1, half = w & 1;
  const int col0 = gate * Hc + j0 + half * 32;
  const int nk = Hc >> 5;     // 16

  if (tid < 32) {  // wave 0: stage h tile + first Wr chunk via TDM
    tdm_load(32768u, h_src + (size_t)m0 * Hc,
             (unsigned)Hc, (unsigned)Bc, 0, (unsigned)Hc, 16, 0, (unsigned)Hc, 0);
    tdm_load(0u, WrT + (size_t)j0 * Hc,
             (unsigned)Hc, (unsigned)G4, 4, 32, 64, 4,
             (unsigned)Hc, (unsigned long long)Hc * (unsigned long long)Hc);
  }

  // accumulator init = xw (C-matrix layout: lanes 0-15 rows 0-7, lanes 16-31 rows 8-15)
  const int n = lane & 15, mb = (lane >> 4) * 8;
  AccU u0, u1;
#pragma unroll
  for (int r = 0; r < 8; ++r) {
    size_t mrow = (size_t)(m0 + mb + r) * ((size_t)Tc * G4);
    u0.f[r] = xw_t[mrow + col0 + n];
    u1.f[r] = xw_t[mrow + col0 + 16 + n];
  }
  v8f acc0 = u0.v, acc1 = u1.v;

  const unsigned short* hA = (const unsigned short*)(smem + 32768);
  for (int i = 0; i < nk; ++i) {
    if (tid < 32) {
      if (i + 1 < nk) {
        tdm_load(16384u * (unsigned)((i + 1) & 1), WrT + (size_t)j0 * Hc + ((i + 1) << 5),
                 (unsigned)Hc, (unsigned)G4, 4, 32, 64, 4,
                 (unsigned)Hc, (unsigned long long)Hc * (unsigned long long)Hc);
        __builtin_amdgcn_s_wait_tensorcnt(1);   // h + current chunk done, next in flight
      } else {
        __builtin_amdgcn_s_wait_tensorcnt(0);
      }
    }
    __syncthreads();
    const unsigned short* Bls = (const unsigned short*)(smem + 16384 * (i & 1));
    v16bf a  = load_frag(hA, Hc, i << 5, lane);
    v16bf b0 = load_frag(&Bls[(gate * 64 + half * 32)      * 32], 32, 0, lane);
    v16bf b1 = load_frag(&Bls[(gate * 64 + half * 32 + 16) * 32], 32, 0, lane);
    acc0 = WMMA_BF16(a, b0, acc0);
    acc1 = WMMA_BF16(a, b1, acc1);
    __syncthreads();
  }

  // park z in LDS (overlay buffer 0; last k-chunk used buffer 1)
  float* zsf = (float*)smem;   // [4][16][64]
  u0.v = acc0; u1.v = acc1;
#pragma unroll
  for (int r = 0; r < 8; ++r) {
    zsf[((gate * 16) + mb + r) * 64 + half * 32 + n]      = u0.f[r];
    zsf[((gate * 16) + mb + r) * 64 + half * 32 + 16 + n] = u1.f[r];
  }
  __syncthreads();

#pragma unroll
  for (int e = 0; e < 4; ++e) {
    int idx = tid + 256 * e;              // 1024 = 16 rows x 64 cols
    int row = idx >> 6, cj = idx & 63;
    int b = m0 + row, j = j0 + cj;
    float iv = sigm(zsf[(0 * 16 + row) * 64 + cj]);
    float fv = sigm(zsf[(1 * 16 + row) * 64 + cj]);
    float gv = tanhx(zsf[(2 * 16 + row) * 64 + cj]);
    float ov = sigm(zsf[(3 * 16 + row) * 64 + cj]);
    size_t cidx = (size_t)b * Hc + j;
    float cn = fv * c[cidx] + iv * gv;
    c[cidx] = cn;
    float h = ov * tanhx(cn);
    seq_t[(size_t)b * ((size_t)Tc * Hc) + j] = h;
    h_dst[cidx] = f2bf(h);
  }
}

// ---------------- merge ave -> bf16 ----------------
__global__ __launch_bounds__(256) void k_merge(const float* __restrict__ a,
                                               const float* __restrict__ b,
                                               unsigned short* __restrict__ o, size_t nn) {
  size_t i = (size_t)blockIdx.x * 256 + threadIdx.x;
  if (i < nn) o[i] = f2bf(0.5f * (a[i] + b[i]));
}

// ---------------- final h,c tails ----------------
__global__ __launch_bounds__(256) void k_tail(const float* __restrict__ seq,
                                              const float* __restrict__ c,
                                              float* __restrict__ hout,
                                              float* __restrict__ cout) {
  int i = blockIdx.x * 256 + threadIdx.x;
  if (i < Bc * Hc) {
    int b = i / Hc, j = i % Hc;
    hout[i] = seq[((size_t)b * Tc + (Tc - 1)) * Hc + j];
    cout[i] = c[i];
  }
}

extern "C" void kernel_launch(void* const* d_in, const int* in_sizes, int n_in,
                              void* d_out, int out_size, void* d_ws, size_t ws_size,
                              hipStream_t stream) {
  const int*   x    = (const int*)  d_in[0];
  const float* emb  = (const float*)d_in[1];
  const float* Wk_f = (const float*)d_in[2];
  const float* Wr_f = (const float*)d_in[3];
  const float* b_f  = (const float*)d_in[4];
  const float* Wk_b = (const float*)d_in[5];
  const float* Wr_b = (const float*)d_in[6];
  const float* b_b  = (const float*)d_in[7];
  const float* Wk2  = (const float*)d_in[8];
  const float* Wr2  = (const float*)d_in[9];
  const float* b2   = (const float*)d_in[10];
  float* out = (float*)d_out;

  char* ws = (char*)d_ws;
  size_t off = 0;
  auto alloc = [&](size_t bytes) { size_t p = off; off += (bytes + 255) & ~(size_t)255; return p; };

  unsigned short* eb     = (unsigned short*)(ws + alloc((size_t)Bc * Tc * Ec * 2));
  unsigned short* WkTf   = (unsigned short*)(ws + alloc((size_t)G4 * Ec * 2));
  unsigned short* WrTf   = (unsigned short*)(ws + alloc((size_t)G4 * Hc * 2));
  unsigned short* WkTb   = (unsigned short*)(ws + alloc((size_t)G4 * Ec * 2));
  unsigned short* WrTb   = (unsigned short*)(ws + alloc((size_t)G4 * Hc * 2));
  unsigned short* WkT2   = (unsigned short*)(ws + alloc((size_t)G4 * Hc * 2));
  unsigned short* WrT2   = (unsigned short*)(ws + alloc((size_t)G4 * Hc * 2));
  float*          xw     = (float*)         (ws + alloc((size_t)Bc * Tc * G4 * 4));
  float*          seqf   = (float*)         (ws + alloc((size_t)Bc * Tc * Hc * 4));
  float*          seqb   = (float*)         (ws + alloc((size_t)Bc * Tc * Hc * 4));
  unsigned short* merged = (unsigned short*)(ws + alloc((size_t)Bc * Tc * Hc * 2));
  unsigned short* hb0    = (unsigned short*)(ws + alloc((size_t)Bc * Hc * 2));
  unsigned short* hb1    = (unsigned short*)(ws + alloc((size_t)Bc * Hc * 2));
  float*          cbuf   = (float*)         (ws + alloc((size_t)Bc * Hc * 4));
  (void)ws_size; (void)in_sizes; (void)n_in; (void)out_size;

  const int nState = Bc * Hc;
  const int gState = (nState + 255) / 256;
  const size_t gemmLds = 24576;   // As[2] 8KB + Bs[2] 16KB
  const size_t stepLds = 49152;   // Bls[2] 32KB + hA 16KB (z overlays Bls0)

  // weight transposes (f32 [K,N] -> bf16 [N,K])
  k_transpose_cast<<<((size_t)Ec * G4 + 255) / 256, 256, 0, stream>>>(Wk_f, WkTf, Ec, G4);
  k_transpose_cast<<<((size_t)Hc * G4 + 255) / 256, 256, 0, stream>>>(Wr_f, WrTf, Hc, G4);
  k_transpose_cast<<<((size_t)Ec * G4 + 255) / 256, 256, 0, stream>>>(Wk_b, WkTb, Ec, G4);
  k_transpose_cast<<<((size_t)Hc * G4 + 255) / 256, 256, 0, stream>>>(Wr_b, WrTb, Hc, G4);
  k_transpose_cast<<<((size_t)Hc * G4 + 255) / 256, 256, 0, stream>>>(Wk2, WkT2, Hc, G4);
  k_transpose_cast<<<((size_t)Hc * G4 + 255) / 256, 256, 0, stream>>>(Wr2, WrT2, Hc, G4);

  // embedding
  k_embed<<<Bc * Tc, 256, 0, stream>>>(x, emb, eb);

  dim3 gemmGrid(G4 / 128, (Bc * Tc) / 64);
  dim3 stepGrid(Bc / 16, Hc / 64);

  // ---- layer 1 forward ----
  k_zero_state<<<gState, 256, 0, stream>>>(hb0, hb1, cbuf, nState);
  k_gemm_bias<<<gemmGrid, 256, gemmLds, stream>>>(eb, WkTf, b_f, xw, Bc * Tc, G4, Ec);
  for (int t = 0; t < Tc; ++t) {
    const unsigned short* hs = (t & 1) ? hb1 : hb0;
    unsigned short*       hd = (t & 1) ? hb0 : hb1;
    k_lstm_step<<<stepGrid, 256, stepLds, stream>>>(xw + (size_t)t * G4, WrTf, hs, hd, cbuf,
                                                    seqf + (size_t)t * Hc);
  }

  // ---- layer 1 backward (iterate time in reverse, write at original t) ----
  k_zero_state<<<gState, 256, 0, stream>>>(hb0, hb1, cbuf, nState);
  k_gemm_bias<<<gemmGrid, 256, gemmLds, stream>>>(eb, WkTb, b_b, xw, Bc * Tc, G4, Ec);
  for (int s = 0; s < Tc; ++s) {
    int t = Tc - 1 - s;
    const unsigned short* hs = (s & 1) ? hb1 : hb0;
    unsigned short*       hd = (s & 1) ? hb0 : hb1;
    k_lstm_step<<<stepGrid, 256, stepLds, stream>>>(xw + (size_t)t * G4, WrTb, hs, hd, cbuf,
                                                    seqb + (size_t)t * Hc);
  }

  // ---- merge ave ----
  k_merge<<<((size_t)Bc * Tc * Hc + 255) / 256, 256, 0, stream>>>(seqf, seqb, merged,
                                                                  (size_t)Bc * Tc * Hc);

  // ---- layer 2 (seq straight into d_out) ----
  k_zero_state<<<gState, 256, 0, stream>>>(hb0, hb1, cbuf, nState);
  k_gemm_bias<<<gemmGrid, 256, gemmLds, stream>>>(merged, WkT2, b2, xw, Bc * Tc, G4, Hc);
  for (int t = 0; t < Tc; ++t) {
    const unsigned short* hs = (t & 1) ? hb1 : hb0;
    unsigned short*       hd = (t & 1) ? hb0 : hb1;
    k_lstm_step<<<stepGrid, 256, stepLds, stream>>>(xw + (size_t)t * G4, WrT2, hs, hd, cbuf,
                                                    out + (size_t)t * Hc);
  }

  // ---- tails: h_T, c_T ----
  k_tail<<<gState, 256, 0, stream>>>(out, cbuf, out + (size_t)Bc * Tc * Hc,
                                     out + (size_t)Bc * Tc * Hc + (size_t)Bc * Hc);
}
